// LSTM_5592047420095
// MI455X (gfx1250) — compile-verified, matrix operand order checked by reading, not run on hardware
//
#include <hip/hip_runtime.h>
#include <hip/hip_bf16.h>

// ============================================================================
// 2-layer LSTM w/ Np=8 closed-loop prediction, CDNA5 (gfx1250) WMMA bf16.
//
//  Phase A: convert weights/inputs fp32->bf16 (weights ~25MB bf16: L2-resident)
//  Phase B: sequential filtering, k=0..95: fused LSTM-cell kernel per layer
//           (WMMA GEMM + bias + LDS gate exchange + activations + c/h update),
//           saving per-k (h1,c1) seeds for the rollouts, plus linear head n=0.
//  Phase C: batched rollout n=1..7 over all 96 samples at once (M=12288).
//  Linear head scatters predictions directly into outMusic; the fill kernel
//  writes the disjoint complement (triangular init + zeros).
//
//  Data movement: tiles staged global->LDS via GLOBAL_LOAD_ASYNC_TO_LDS_*
//  (ASYNCcnt + s_wait_asynccnt) when the toolchain exposes the builtins;
//  otherwise direct short-live-range copies (no VGPR staging => no spills).
//
//  Workspace requirement ~231 MB.
// ============================================================================

typedef unsigned short u16;
typedef __attribute__((ext_vector_type(16))) __bf16 v16bf;
typedef __attribute__((ext_vector_type(8)))  float  v8f;
typedef int int4v __attribute__((ext_vector_type(4)));
typedef int int2v __attribute__((ext_vector_type(2)));

union FragBF {
  uint4 u4[2];
  v16bf v;
};

#define NPRED 8
#define HID   1024
#define FEATD 128
#define NSAMP 96
#define NB    128
#define MROLL (NSAMP * NB)  /* 12288 */

#if defined(__AMDGCN__) &&                                        \
    __has_builtin(__builtin_amdgcn_global_load_async_to_lds_b128) && \
    __has_builtin(__builtin_amdgcn_global_load_async_to_lds_b64)  && \
    __has_builtin(__builtin_amdgcn_s_wait_asynccnt)
#define USE_ASYNC_LDS 1
#else
#define USE_ASYNC_LDS 0
#endif

__device__ __forceinline__ u16 f2bf(float f) {
  unsigned u = __float_as_uint(f);
  u += 0x7FFFu + ((u >> 16) & 1u);   // round-to-nearest-even
  return (u16)(u >> 16);
}
__device__ __forceinline__ float sigm(float x) { return 1.0f / (1.0f + __expf(-x)); }

// 16B / 8B global -> LDS copies (async when available: no VGPR staging).
__device__ __forceinline__ void cp_lds16(u16* l, const u16* g) {
#if USE_ASYNC_LDS
  __builtin_amdgcn_global_load_async_to_lds_b128(
      (__attribute__((address_space(1))) int4v*)g,
      (__attribute__((address_space(3))) int4v*)l, 0, 0);
#else
  *(uint4*)l = *(const uint4*)g;
#endif
}
__device__ __forceinline__ void cp_lds8(u16* l, const u16* g) {
#if USE_ASYNC_LDS
  __builtin_amdgcn_global_load_async_to_lds_b64(
      (__attribute__((address_space(1))) int2v*)g,
      (__attribute__((address_space(3))) int2v*)l, 0, 0);
#else
  *(ushort4*)l = *(const ushort4*)g;
#endif
}
__device__ __forceinline__ void wait_stage() {
#if USE_ASYNC_LDS
  __builtin_amdgcn_s_wait_asynccnt(0);
#endif
}

// ---------------------------------------------------------------------------
// Fused LSTM cell: gates = X@Wih^T + Hin@Whh^T + bih + bhh, then c/h update.
// Block = 256 threads (8 waves), tile = 32 rows x 64 hidden units
// (=> 4x 32x64 gate tiles: i,f,g,o; wave w -> gate q=w&3, row-half rh=w>>2).
// ---------------------------------------------------------------------------
__global__ __launch_bounds__(256, 1)
void k_cell(const u16* __restrict__ X, int ldx,          // [M, ldx] bf16
            const u16* __restrict__ Hin,                 // [M, 1024] bf16
            const u16* __restrict__ Wih,                 // [4096, ldx] bf16
            const u16* __restrict__ Whh,                 // [4096, 1024] bf16
            const float* __restrict__ bih, const float* __restrict__ bhh,
            const float* Cin, float* Cout,               // may alias (elementwise)
            u16* __restrict__ Hout,                      // [M, 1024] bf16
            float* CoutCommit, u16* HoutCommitB, float* HoutCommitF) // nullable
{
  __shared__ __align__(16) u16 Alds[2][32][40];        // 32x32 bf16 + pad
  __shared__ __align__(16) u16 Blds[2][4][64][40];     // 4 gates x 64 cols x 32 k
  // gate exchange buffer aliases B double-buffer (used only after GEMM)
  float (*Glds)[32][68] = reinterpret_cast<float (*)[32][68]>(&Blds[0][0][0][0]);

  const int t     = threadIdx.x;
  const int lane  = t & 31, w = t >> 5;
  const int q     = w & 3,  rh = w >> 2;
  const int khalf = lane >> 4, lm = lane & 15;
  const long m0   = (long)blockIdx.x * 32;
  const int  j0   = blockIdx.y * 64;

  const int arow = t >> 3;           // A stage: 4 bf16 (8B) per thread
  const int acol = (t & 7) * 4;
  const int bq   = t >> 6;           // B stage: one 32-elem W row per thread
  const int bn   = t & 63;

  const int nkx = ldx >> 5;
  const int nch = nkx + (HID >> 5);

  v8f acc[4] = {};

  auto stage = [&](int buf, int ic) {
    const u16* Ab; int lda; const u16* Bb; int ldb; int kb;
    if (ic < nkx) { Ab = X;   lda = ldx; Bb = Wih; ldb = ldx; kb = ic << 5; }
    else          { Ab = Hin; lda = HID; Bb = Whh; ldb = HID; kb = (ic - nkx) << 5; }
    cp_lds8(&Alds[buf][arow][acol], Ab + (m0 + arow) * (long)lda + kb + acol);
    const u16* brow = Bb + ((long)(bq * HID + j0 + bn)) * ldb + kb;
    u16* bl = &Blds[buf][bq][bn][0];
    cp_lds16(bl + 0,  brow + 0);
    cp_lds16(bl + 8,  brow + 8);
    cp_lds16(bl + 16, brow + 16);
    cp_lds16(bl + 24, brow + 24);
  };

  stage(0, 0);
  int buf = 0;
  for (int ic = 0; ic < nch; ++ic) {
    wait_stage();
    __syncthreads();
    if (ic + 1 < nch) stage(buf ^ 1, ic + 1);   // async copy overlaps WMMA below
    FragBF a, bfr[4];
    a.u4[0] = *(const uint4*)&Alds[buf][rh * 16 + lm][khalf * 8];
    a.u4[1] = *(const uint4*)&Alds[buf][rh * 16 + lm][16 + khalf * 8];
#pragma unroll
    for (int ct = 0; ct < 4; ++ct) {
      bfr[ct].u4[0] = *(const uint4*)&Blds[buf][q][ct * 16 + lm][khalf * 8];
      bfr[ct].u4[1] = *(const uint4*)&Blds[buf][q][ct * 16 + lm][16 + khalf * 8];
    }
#pragma unroll
    for (int ct = 0; ct < 4; ++ct) {            // back-to-back WMMA issue
      acc[ct] = __builtin_amdgcn_wmma_f32_16x16x32_bf16(
          false, a.v, false, bfr[ct].v, (short)0, acc[ct], false, false);
    }
    buf ^= 1;
  }

  __syncthreads();   // all B reads done before gate buffer aliases it
#pragma unroll
  for (int ct = 0; ct < 4; ++ct) {
    int nl = ct * 16 + lm;
    float bsum = bih[q * HID + j0 + nl] + bhh[q * HID + j0 + nl];
#pragma unroll
    for (int v = 0; v < 8; ++v) {
      int ml = rh * 16 + v + (khalf ? 8 : 0);
      Glds[q][ml][nl] = acc[ct][v] + bsum;
    }
  }
  __syncthreads();

  for (int idx = t; idx < 32 * 64; idx += 256) {
    int m = idx >> 6, n = idx & 63;
    long o = (m0 + m) * (long)HID + (j0 + n);
    float gi = Glds[0][m][n], gf = Glds[1][m][n];
    float gg = Glds[2][m][n], go = Glds[3][m][n];
    float cold = Cin[o];
    float c2 = sigm(gf) * cold + sigm(gi) * tanhf(gg);
    float h2 = sigm(go) * tanhf(c2);
    Cout[o] = c2;
    Hout[o] = f2bf(h2);
    if (CoutCommit)  CoutCommit[o]  = c2;
    if (HoutCommitB) HoutCommitB[o] = f2bf(h2);
    if (HoutCommitF) HoutCommitF[o] = h2;
  }
}

// ---------------------------------------------------------------------------
// Linear head: pred = Top@Wlin^T + blin. Scatters valid predictions straight
// into outMusic[k+n+1, b, f, n] (valid iff r>=2n+2 && r<96), always writes
// bf16 feedback FB for the next closed-loop step, optional nextSampleEst.
// Block = 256 threads, tile = 32 rows x 128 cols.
// ---------------------------------------------------------------------------
__global__ __launch_bounds__(256, 1)
void k_lin(const u16* __restrict__ Top,      // [M, 1024] bf16
           const u16* __restrict__ Wlin,     // [128, 1024] bf16
           const float* __restrict__ blin,   // [128]
           float* __restrict__ outMusic,     // [96,128,128,8]
           u16* __restrict__ FB,             // [M, 128] bf16
           float* nextEst,                   // nullable [128,128]
           int n, int k0)
{
  __shared__ __align__(16) u16 Alds[2][32][40];
  __shared__ __align__(16) u16 Blds[2][128][40];

  const int t     = threadIdx.x;
  const int lane  = t & 31, w = t >> 5;
  const int rh    = w >> 2, cbase = (w & 3) * 32;
  const int khalf = lane >> 4, lm = lane & 15;
  const long m0   = (long)blockIdx.x * 32;

  const int arow = t >> 3, acol = (t & 7) * 4;
  const int brow = t >> 1, bhalf = (t & 1) * 16;

  v8f acc[2] = {};

  auto stage = [&](int buf, int ic) {
    int kb = ic << 5;
    cp_lds8(&Alds[buf][arow][acol], Top + (m0 + arow) * (long)HID + kb + acol);
    const u16* bp = Wlin + (long)brow * HID + kb + bhalf;
    u16* bl = &Blds[buf][brow][bhalf];
    cp_lds16(bl + 0, bp + 0);
    cp_lds16(bl + 8, bp + 8);
  };

  stage(0, 0);
  int buf = 0;
  for (int ic = 0; ic < (HID >> 5); ++ic) {
    wait_stage();
    __syncthreads();
    if (ic + 1 < (HID >> 5)) stage(buf ^ 1, ic + 1);
    FragBF a, bfr[2];
    a.u4[0] = *(const uint4*)&Alds[buf][rh * 16 + lm][khalf * 8];
    a.u4[1] = *(const uint4*)&Alds[buf][rh * 16 + lm][16 + khalf * 8];
#pragma unroll
    for (int ct = 0; ct < 2; ++ct) {
      int bnc = cbase + ct * 16 + lm;
      bfr[ct].u4[0] = *(const uint4*)&Blds[buf][bnc][khalf * 8];
      bfr[ct].u4[1] = *(const uint4*)&Blds[buf][bnc][16 + khalf * 8];
    }
#pragma unroll
    for (int ct = 0; ct < 2; ++ct) {
      acc[ct] = __builtin_amdgcn_wmma_f32_16x16x32_bf16(
          false, a.v, false, bfr[ct].v, (short)0, acc[ct], false, false);
    }
    buf ^= 1;
  }

#pragma unroll
  for (int ct = 0; ct < 2; ++ct) {
    int ncol = cbase + ct * 16 + lm;
    float bl = blin[ncol];
#pragma unroll
    for (int v = 0; v < 8; ++v) {
      long mrow = m0 + rh * 16 + v + (khalf ? 8 : 0);
      float val = acc[ct][v] + bl;
      int k = k0 + (int)(mrow >> 7);
      int b = (int)(mrow & 127);
      int r = k + n + 1;
      if (r >= 2 * n + 2 && r < NSAMP)
        outMusic[(((long)r * NB + b) * FEATD + ncol) * NPRED + n] = val;
      FB[mrow * (long)FEATD + ncol] = f2bf(val);
      if (nextEst) nextEst[(long)b * FEATD + ncol] = val;
    }
  }
}

// ---------------------------------------------------------------------------
__global__ void k_f2b(const float* __restrict__ s, u16* __restrict__ d, long n) {
  long i  = (long)blockIdx.x * blockDim.x + threadIdx.x;
  long st = (long)gridDim.x * blockDim.x;
  for (; i < n; i += st) d[i] = f2bf(s[i]);
}
__global__ void k_copy(const float* __restrict__ s, float* __restrict__ d, long n) {
  long i  = (long)blockIdx.x * blockDim.x + threadIdx.x;
  long st = (long)gridDim.x * blockDim.x;
  for (; i < n; i += st) d[i] = s[i];
}
// Writes the complement of the scatter region: tri init (r<8, n>=r) + zeros.
__global__ void k_fill(const float* __restrict__ inMusic, float* __restrict__ outMusic) {
  const long total = (long)NSAMP * NB * FEATD * NPRED;
  long i  = (long)blockIdx.x * blockDim.x + threadIdx.x;
  long st = (long)gridDim.x * blockDim.x;
  for (; i < total; i += st) {
    int n = (int)(i & 7);
    long rest = i >> 3;
    int f = (int)(rest & 127); rest >>= 7;
    int b = (int)(rest & 127);
    int r = (int)(rest >> 7);
    if (r < 2 * n + 2) {
      float v = 0.0f;
      if (r < NPRED && n >= r) v = inMusic[((long)r * NB + b) * FEATD + f];
      outMusic[i] = v;
    }
  }
}

// ---------------------------------------------------------------------------
extern "C" void kernel_launch(void* const* d_in, const int* in_sizes, int n_in,
                              void* d_out, int out_size, void* d_ws, size_t ws_size,
                              hipStream_t stream) {
  (void)in_sizes; (void)n_in; (void)out_size; (void)ws_size;

  const float* inMusic = (const float*)d_in[0];
  const float* h0      = (const float*)d_in[1];
  const float* c0      = (const float*)d_in[2];
  const float* Wih0    = (const float*)d_in[3];
  const float* Whh0    = (const float*)d_in[4];
  const float* bih0    = (const float*)d_in[5];
  const float* bhh0    = (const float*)d_in[6];
  const float* Wih1    = (const float*)d_in[7];
  const float* Whh1    = (const float*)d_in[8];
  const float* bih1    = (const float*)d_in[9];
  const float* bhh1    = (const float*)d_in[10];
  const float* Wlin    = (const float*)d_in[11];
  const float* blin    = (const float*)d_in[12];
  float* out = (float*)d_out;

  float* outMusic = out;
  float* outHF    = out + (long)NSAMP * NB * FEATD * NPRED;  // 12,582,912
  float* outCF    = outHF + 2 * NB * HID;                    // +262,144
  float* outNext  = outCF + 2 * NB * HID;                    // +262,144

  // ---- carve workspace (~231 MB) ----
  size_t off = 0;
  auto carve = [&](size_t bytes) -> void* {
    void* p = (char*)d_ws + off;
    off += (bytes + 255) & ~(size_t)255;
    return p;
  };
  const size_t SL = (size_t)NB * HID;        // 131072: one (k, layer) h/c slice
  const size_t L1 = (size_t)NSAMP * SL;      // 12,582,912: layer stride

  u16* Wih0b = (u16*)carve((size_t)4096 * 128  * 2);
  u16* Whh0b = (u16*)carve((size_t)4096 * 1024 * 2);
  u16* Wih1b = (u16*)carve((size_t)4096 * 1024 * 2);
  u16* Whh1b = (u16*)carve((size_t)4096 * 1024 * 2);
  u16* Wlinb = (u16*)carve((size_t)128  * 1024 * 2);
  u16* Xb    = (u16*)carve((size_t)NSAMP * NB * FEATD * 2);
  u16* HFbA  = (u16*)carve(2 * SL * 2);      // committed h, ping
  u16* HFbB  = (u16*)carve(2 * SL * 2);      // committed h, pong
  u16* FBb   = (u16*)carve((size_t)MROLL * FEATD * 2);
  u16* Hp0   = (u16*)carve(2 * L1 * 2);      // per-k h states, ping
  u16* Hp1   = (u16*)carve(2 * L1 * 2);      // per-k h states, pong
  float* CF    = (float*)carve(2 * SL * 4);  // committed c (fp32, also cell input)
  float* HFf   = (float*)carve(2 * SL * 4);  // committed h (fp32, for output)
  float* Csave = (float*)carve(2 * L1 * 4);  // per-k c states (fp32, in-place rollout)

  // ---- Phase A: conversions / init ----
  k_f2b<<<1024, 256, 0, stream>>>(Wih0, Wih0b, (long)4096 * 128);
  k_f2b<<<1024, 256, 0, stream>>>(Whh0, Whh0b, (long)4096 * 1024);
  k_f2b<<<1024, 256, 0, stream>>>(Wih1, Wih1b, (long)4096 * 1024);
  k_f2b<<<1024, 256, 0, stream>>>(Whh1, Whh1b, (long)4096 * 1024);
  k_f2b<<<256,  256, 0, stream>>>(Wlin, Wlinb, (long)128 * 1024);
  k_f2b<<<1024, 256, 0, stream>>>(inMusic, Xb, (long)NSAMP * NB * FEATD);
  k_f2b<<<256,  256, 0, stream>>>(h0, HFbA, (long)2 * SL);
  k_f2b<<<256,  256, 0, stream>>>(h0, HFbB, (long)2 * SL);
  k_copy<<<256, 256, 0, stream>>>(h0, HFf, (long)2 * SL);
  k_copy<<<256, 256, 0, stream>>>(c0, CF,  (long)2 * SL);

  // tri-init + zeros (disjoint from prediction scatter region)
  k_fill<<<8192, 256, 0, stream>>>(inMusic, outMusic);

  // ---- Phase B: sequential filtering, k = 0..95 ----
  for (int k = 0; k < NSAMP; ++k) {
    bool cm = (k >= 1);                    // hF/cF commit gated on k>=1
    u16* HFrd = (k & 1) ? HFbB : HFbA;     // read/commit ping-pong (GEMM reads
    u16* HFwr = (k & 1) ? HFbA : HFbB;     //  all columns -> no in-place write)
    dim3 grid(NB / 32, HID / 64);

    // layer 0
    k_cell<<<grid, 256, 0, stream>>>(
        Xb + (size_t)k * NB * FEATD, FEATD,
        HFrd + 0,
        Wih0b, Whh0b, bih0, bhh0,
        CF + 0,
        Csave + 0 * L1 + (size_t)k * SL,
        Hp0   + 0 * L1 + (size_t)k * SL,
        cm ? CF   + 0 : nullptr,
        cm ? HFwr + 0 : nullptr,
        cm ? HFf  + 0 : nullptr);

    // layer 1 (input = layer-0 h just written)
    k_cell<<<grid, 256, 0, stream>>>(
        Hp0 + 0 * L1 + (size_t)k * SL, HID,
        HFrd + SL,
        Wih1b, Whh1b, bih1, bhh1,
        CF + SL,
        Csave + 1 * L1 + (size_t)k * SL,
        Hp0   + 1 * L1 + (size_t)k * SL,
        cm ? CF   + SL : nullptr,
        cm ? HFwr + SL : nullptr,
        cm ? HFf  + SL : nullptr);

    // linear head, n=0 (scatter + feedback seed; k=95 -> nextSampleEst)
    k_lin<<<dim3(NB / 32), 256, 0, stream>>>(
        Hp0 + 1 * L1 + (size_t)k * SL,
        Wlinb, blin, outMusic,
        FBb + (size_t)k * NB * FEATD,
        (k == NSAMP - 1) ? outNext : nullptr,
        /*n=*/0, /*k0=*/k);
  }

  // final committed hF/cF -> outputs (rollout never touches HFf/CF)
  k_copy<<<256, 256, 0, stream>>>(HFf, outHF, (long)2 * SL);
  k_copy<<<256, 256, 0, stream>>>(CF,  outCF, (long)2 * SL);

  // ---- Phase C: batched closed-loop rollout, n = 1..7, M = 12288 ----
  int p = 0;
  for (int n = 1; n < NPRED; ++n) {
    u16* Hrd = (p == 0) ? Hp0 : Hp1;
    u16* Hwr = (p == 0) ? Hp1 : Hp0;
    dim3 grid(MROLL / 32, HID / 64);

    // layer 0: x = previous prediction (FB), c updated in place (elementwise)
    k_cell<<<grid, 256, 0, stream>>>(
        FBb, FEATD,
        Hrd + 0 * L1,
        Wih0b, Whh0b, bih0, bhh0,
        Csave + 0 * L1, Csave + 0 * L1,
        Hwr + 0 * L1,
        nullptr, nullptr, nullptr);

    // layer 1
    k_cell<<<grid, 256, 0, stream>>>(
        Hwr + 0 * L1, HID,
        Hrd + 1 * L1,
        Wih1b, Whh1b, bih1, bhh1,
        Csave + 1 * L1, Csave + 1 * L1,
        Hwr + 1 * L1,
        nullptr, nullptr, nullptr);

    // linear head: scatter preds[k, n] into outMusic + next feedback
    k_lin<<<dim3(MROLL / 32), 256, 0, stream>>>(
        Hwr + 1 * L1,
        Wlinb, blin, outMusic,
        FBb, nullptr,
        /*n=*/n, /*k0=*/0);

    p ^= 1;
  }
}
